// imamba_77653008711868
// MI455X (gfx1250) — compile-verified, matrix operand order checked by reading, not run on hardware
//
#include <hip/hip_runtime.h>
#include <hip/hip_bf16.h>
#include <math.h>

// ---------------------------------------------------------------------------
// iMamba forward (bidirectional Mamba, 2 layers) for MI455X / gfx1250.
// GEMMs: f16 operands pre-converted in global, tiles staged to LDS via
// GLOBAL_LOAD_ASYNC_TO_LDS_B128 (double-buffered, ASYNCcnt-pipelined) when
// the builtin is available, then v_wmma_f32_16x16x32_f16 (f32 accum).
// Selective scan in f32 VALU with LDS-staged B/C + global prefetch.
// ---------------------------------------------------------------------------

#define BATCH    16
#define NUMLEN   96
#define PREDLEN  192
#define SEQL     1024
#define DMODEL   512
#define DSTATE   16
#define DCONV    4
#define DINNER   1024
#define DTRANK   32
#define NLAYERS  2
#define ROWS     (BATCH * SEQL)          // 16384 GEMM rows

typedef __attribute__((ext_vector_type(16))) _Float16 v16h;
typedef __attribute__((ext_vector_type(8)))  float    v8f;
typedef __attribute__((ext_vector_type(4)))  int      v4i;

#define GLOBAL_AS __attribute__((address_space(1)))
#define LDS_AS    __attribute__((address_space(3)))

#if defined(__has_builtin)
#  if __has_builtin(__builtin_amdgcn_global_load_async_to_lds_b128)
#    define HAS_ASYNC_LDS 1
#  endif
#endif
#ifndef HAS_ASYNC_LDS
#  define HAS_ASYNC_LDS 0
#endif

#if defined(__has_builtin) && __has_builtin(__builtin_amdgcn_s_wait_asynccnt)
#  define WAIT_ASYNC(n) __builtin_amdgcn_s_wait_asynccnt(n)
#else
#  define WAIT_ASYNC(n) asm volatile("s_wait_asynccnt " #n ::: "memory")
#endif

// generic -> addrspace'd v4i* (two-step: addrspacecast, then pointee cast)
#define AS_GLOBAL_V4I(p) ((GLOBAL_AS v4i*)(GLOBAL_AS void*)(void*)(p))
#define AS_LDS_V4I(p)    ((LDS_AS    v4i*)(LDS_AS    void*)(void*)(p))

__device__ __forceinline__ float siluf(float x) { return x / (1.0f + __expf(-x)); }
__device__ __forceinline__ float softplusf(float x) {
    return (x > 20.0f) ? x : log1pf(__expf(x));
}

// ---------------------------------------------------------------------------
// WMMA GEMM: C[M,N] = A[M,K] * W[N,K]^T (+ bias[n]) (+ softplus)
// A, W are f16 in global. M%64==0, N%64==0, K%32==0, 16B-aligned rows.
// Block: 128 threads = 4 waves (2x2); wave tile 32x32 = 4 WMMA accumulators.
// mode: 0 = none, 1 = +bias, 2 = +bias then softplus
// ---------------------------------------------------------------------------
__global__ __launch_bounds__(128)
void gemm_wmma_kernel(const _Float16* __restrict__ A, int lda,
                      const _Float16* __restrict__ W, int ldw,
                      const float* __restrict__ bias,
                      float* __restrict__ C, int ldc,
                      int K, int mode) {
    __shared__ __align__(16) _Float16 sA[2][64 * 32];
    __shared__ __align__(16) _Float16 sW[2][64 * 32];

    const int tid  = threadIdx.x;
    const int lane = tid & 31;
    const int wid  = tid >> 5;          // 0..3
    const int wm   = wid >> 1;          // wave row (0..1)
    const int wn   = wid & 1;           // wave col (0..1)
    const int rowBase = blockIdx.y * 64;
    const int colBase = blockIdx.x * 64;
    const int ktiles  = K >> 5;

    v8f acc[2][2] = {};

#if HAS_ASYNC_LDS
    // Async staging: each thread copies 2 x b128 chunks of A and of W per tile
    // (tile = 64 rows x 32 halves = 4KB = 256 chunks of 16B).
    auto issue_tile = [&](int kt, int buf) {
        #pragma unroll
        for (int h = 0; h < 2; ++h) {
            const int c   = tid + h * 128;      // 0..255
            const int row = c >> 2;
            const int q   = (c & 3) * 8;        // halves within 32-half row
            const _Float16* gA = A + (size_t)(rowBase + row) * lda + kt * 32 + q;
            const _Float16* gW = W + (size_t)(colBase + row) * ldw + kt * 32 + q;
            __builtin_amdgcn_global_load_async_to_lds_b128(
                AS_GLOBAL_V4I(gA), AS_LDS_V4I(&sA[buf][row * 32 + q]), 0, 0);
            __builtin_amdgcn_global_load_async_to_lds_b128(
                AS_GLOBAL_V4I(gW), AS_LDS_V4I(&sW[buf][row * 32 + q]), 0, 0);
        }
    };
    issue_tile(0, 0);
#endif

    for (int k = 0; k < ktiles; ++k) {
        int cur;
#if HAS_ASYNC_LDS
        cur = k & 1;
        if (k + 1 < ktiles) {           // prefetch next tile into other buffer
            issue_tile(k + 1, (k + 1) & 1);
            WAIT_ASYNC(4);              // 4 in-flight per thread = next tile only
        } else {
            WAIT_ASYNC(0);
        }
        __syncthreads();                // all waves' tile-k data visible
#else
        cur = 0;
        __syncthreads();                // protect previous iteration's reads
        {   // synchronous staging from f16 global (2 x 16B per matrix/thread)
            #pragma unroll
            for (int h = 0; h < 2; ++h) {
                const int c   = tid + h * 128;
                const int row = c >> 2;
                const int q   = (c & 3) * 8;
                const uint4* gA = reinterpret_cast<const uint4*>(
                    A + (size_t)(rowBase + row) * lda + k * 32 + q);
                const uint4* gW = reinterpret_cast<const uint4*>(
                    W + (size_t)(colBase + row) * ldw + k * 32 + q);
                *reinterpret_cast<uint4*>(&sA[0][row * 32 + q]) = *gA;
                *reinterpret_cast<uint4*>(&sW[0][row * 32 + q]) = *gW;
            }
        }
        __syncthreads();
#endif

        // CDNA5 16-bit fragment striping: lane holds row (lane&15);
        // K elements {kb..kb+7, kb+16..kb+23}, kb = (lane>>4)*8.
        const int fr = lane & 15;
        const int kb = (lane >> 4) * 8;

        v16h af[2], bf[2];
        #pragma unroll
        for (int ti = 0; ti < 2; ++ti) {
            const _Float16* p = &sA[cur][(wm * 32 + ti * 16 + fr) * 32 + kb];
            #pragma unroll
            for (int j = 0; j < 8; ++j) { af[ti][j] = p[j]; af[ti][8 + j] = p[16 + j]; }
        }
        #pragma unroll
        for (int tj = 0; tj < 2; ++tj) {
            const _Float16* p = &sW[cur][(wn * 32 + tj * 16 + fr) * 32 + kb];
            #pragma unroll
            for (int j = 0; j < 8; ++j) { bf[tj][j] = p[j]; bf[tj][8 + j] = p[16 + j]; }
        }

        #pragma unroll
        for (int ti = 0; ti < 2; ++ti)
            #pragma unroll
            for (int tj = 0; tj < 2; ++tj)
                acc[ti][tj] = __builtin_amdgcn_wmma_f32_16x16x32_f16(
                    false, af[ti], false, bf[tj], (short)0, acc[ti][tj], false, false);

#if HAS_ASYNC_LDS
        __syncthreads();                // all reads of buf[cur] done before reuse
#endif
    }

    // epilogue: C/D layout — lane 0-15: M=e, N=lane; lane 16-31: M=8+e, N=lane-16
    const int mOff = (lane >> 4) * 8;
    const int nOff = lane & 15;
    #pragma unroll
    for (int ti = 0; ti < 2; ++ti) {
        #pragma unroll
        for (int tj = 0; tj < 2; ++tj) {
            const int nGlob = colBase + wn * 32 + tj * 16 + nOff;
            #pragma unroll
            for (int e = 0; e < 8; ++e) {
                const int mGlob = rowBase + wm * 32 + ti * 16 + mOff + e;
                float v = acc[ti][tj][e];
                if (mode >= 1) v += bias[nGlob];
                if (mode == 2) v = softplusf(v);
                C[(size_t)mGlob * ldc + nGlob] = v;
            }
        }
    }
}

// ---------------------------------------------------------------------------
// Elementwise / data-movement kernels
// ---------------------------------------------------------------------------

// f32 -> f16 dense convert; count % 1024 == 0; 4 elements per thread
__global__ void f32_to_f16_kernel(const float* __restrict__ in, _Float16* __restrict__ out) {
    int i = (blockIdx.x * blockDim.x + threadIdx.x) * 4;
    float4 f = *reinterpret_cast<const float4*>(in + i);
    _Float16* o = out + i;
    o[0] = (_Float16)f.x; o[1] = (_Float16)f.y;
    o[2] = (_Float16)f.z; o[3] = (_Float16)f.w;
}

// inputs (B, NUMLEN, L) -> f16 A-operand ((B*L), NUMLEN)
__global__ void transpose_in_f16_kernel(const float* __restrict__ in, _Float16* __restrict__ out) {
    int idx = blockIdx.x * blockDim.x + threadIdx.x;   // over ROWS*NUMLEN
    int n   = idx % NUMLEN;
    int rl  = idx / NUMLEN;
    int l   = rl & (SEQL - 1);
    int b   = rl >> 10;
    out[idx] = (_Float16)in[(size_t)b * NUMLEN * SEQL + (size_t)n * SEQL + l];
}

// depthwise causal conv(4) + bias + SiLU on x-half of xz
__global__ void conv_silu_kernel(const float* __restrict__ xz,
                                 const float* __restrict__ cw,
                                 const float* __restrict__ cb,
                                 float* __restrict__ out) {
    int idx = blockIdx.x * blockDim.x + threadIdx.x;   // over ROWS*DINNER
    int d   = idx & (DINNER - 1);
    int row = idx >> 10;
    int l   = row & (SEQL - 1);
    int base = row - l;                                // b*SEQL
    float acc = cb[d];
    #pragma unroll
    for (int k = 0; k < DCONV; ++k) {
        int ls = l - (DCONV - 1) + k;
        if (ls >= 0)
            acc += xz[((size_t)(base + ls) << 11) + d] * cw[d * DCONV + k];
    }
    out[idx] = siluf(acc);
}

// selective scan: one thread per (b, d); 16-state recurrence over L steps
__global__ __launch_bounds__(256)
void scan_kernel(const float* __restrict__ dtb,
                 const float* __restrict__ xc,
                 const float* __restrict__ xdbl,
                 const float* __restrict__ alog,
                 const float* __restrict__ dskip,
                 float* __restrict__ y) {
    __shared__ float sB[DSTATE];
    __shared__ float sC[DSTATE];

    const int b = blockIdx.y;
    const int d = blockIdx.x * 256 + threadIdx.x;

    float Av[DSTATE];
    #pragma unroll
    for (int s = 0; s < DSTATE; ++s) Av[s] = -__expf(alog[d * DSTATE + s]);
    const float Dp = dskip[d];

    float h[DSTATE];
    #pragma unroll
    for (int s = 0; s < DSTATE; ++s) h[s] = 0.0f;

    for (int t = 0; t < SEQL; ++t) {
        const int row = b * SEQL + t;
        __syncthreads();
        if (threadIdx.x < DSTATE)
            sB[threadIdx.x] = xdbl[(size_t)row * 64 + DTRANK + threadIdx.x];
        else if (threadIdx.x < 2 * DSTATE)
            sC[threadIdx.x - DSTATE] =
                xdbl[(size_t)row * 64 + DTRANK + DSTATE + (threadIdx.x - DSTATE)];
        __syncthreads();

        const float dt  = dtb[(size_t)row * DINNER + d];
        const float xv  = xc [(size_t)row * DINNER + d];
        const float dtx = dt * xv;
        if (t + 1 < SEQL) {
            __builtin_prefetch(&dtb[(size_t)(row + 1) * DINNER + d], 0, 1);
            __builtin_prefetch(&xc [(size_t)(row + 1) * DINNER + d], 0, 1);
        }
        float yv = 0.0f;
        #pragma unroll
        for (int s = 0; s < DSTATE; ++s) {
            h[s] = __expf(dt * Av[s]) * h[s] + dtx * sB[s];
            yv  += h[s] * sC[s];
        }
        y[(size_t)row * DINNER + d] = yv + xv * Dp;
    }
}

// gated = y * silu(z-half of xz)
__global__ void gate_kernel(const float* __restrict__ y,
                            const float* __restrict__ xz,
                            float* __restrict__ out) {
    int idx = blockIdx.x * blockDim.x + threadIdx.x;   // over ROWS*DINNER
    int d   = idx & (DINNER - 1);
    int row = idx >> 10;
    float z = xz[((size_t)row << 11) + DINNER + d];
    out[idx] = y[idx] * siluf(z);
}

// sequence flip along L, f32 -> f16 (direct GEMM A-operand)
__global__ void flip_f16_kernel(const float* __restrict__ in, _Float16* __restrict__ out) {
    int idx = blockIdx.x * blockDim.x + threadIdx.x;   // over ROWS*(DMODEL/4)
    int q   = idx & (DMODEL / 4 - 1);
    int row = idx / (DMODEL / 4);
    int l   = row & (SEQL - 1);
    int b   = row >> 10;
    float4 f = reinterpret_cast<const float4*>(
        in + (size_t)(b * SEQL + (SEQL - 1 - l)) * DMODEL)[q];
    _Float16* o = out + (size_t)row * DMODEL + q * 4;
    o[0] = (_Float16)f.x; o[1] = (_Float16)f.y;
    o[2] = (_Float16)f.z; o[3] = (_Float16)f.w;
}

// emb = silu(r1 + flip(r2))
__global__ void combine_kernel(const float* __restrict__ r1,
                               const float* __restrict__ r2,
                               float* __restrict__ emb) {
    int idx = blockIdx.x * blockDim.x + threadIdx.x;   // over ROWS*DMODEL
    int c   = idx & (DMODEL - 1);
    int row = idx >> 9;
    int l   = row & (SEQL - 1);
    int b   = row >> 10;
    float v = r1[idx] + r2[(size_t)(b * SEQL + (SEQL - 1 - l)) * DMODEL + c];
    emb[idx] = siluf(v);
}

// ftmp ((B*L), PREDLEN) -> out (B, PREDLEN, L)
__global__ void transpose_out_kernel(const float* __restrict__ in, float* __restrict__ out) {
    int idx = blockIdx.x * blockDim.x + threadIdx.x;   // over B*PREDLEN*SEQL
    int l   = idx & (SEQL - 1);
    int pb  = idx >> 10;
    int p   = pb % PREDLEN;
    int b   = pb / PREDLEN;
    out[idx] = in[(size_t)(b * SEQL + l) * PREDLEN + p];
}

// ---------------------------------------------------------------------------
// Host orchestration
// ---------------------------------------------------------------------------
extern "C" void kernel_launch(void* const* d_in, const int* in_sizes, int n_in,
                              void* d_out, int out_size, void* d_ws, size_t ws_size,
                              hipStream_t stream) {
    const float* inputs = (const float*)d_in[0];
    const float* w1     = (const float*)d_in[1];
    const float* b1     = (const float*)d_in[2];
    const float* w_in   = (const float*)d_in[3];
    const float* conv_w = (const float*)d_in[4];
    const float* conv_b = (const float*)d_in[5];
    const float* w_xp   = (const float*)d_in[6];
    const float* w_dt   = (const float*)d_in[7];
    const float* b_dt   = (const float*)d_in[8];
    const float* a_log  = (const float*)d_in[9];
    const float* d_skip = (const float*)d_in[10];
    const float* w_out  = (const float*)d_in[11];
    const float* w2     = (const float*)d_in[12];
    const float* b2     = (const float*)d_in[13];
    float* out = (float*)d_out;

    // workspace layout (floats), f16 operand buffers at the end (~489 MB total)
    float* ws   = (float*)d_ws;
    float* emb  = ws;                                   // 16384*512
    float* xz   = emb  + (size_t)ROWS * DMODEL;         // 16384*2048
    float* xcv  = xz   + (size_t)ROWS * 2 * DINNER;     // 16384*1024
    float* xdbl = xcv  + (size_t)ROWS * DINNER;         // 16384*64
    float* dtb  = xdbl + (size_t)ROWS * 64;             // 16384*1024
    float* ybuf = dtb  + (size_t)ROWS * DINNER;         // 16384*1024
    float* r1   = ybuf + (size_t)ROWS * DINNER;         // 16384*512
    float* r2   = r1   + (size_t)ROWS * DMODEL;         // 16384*512
    float* ftmp = r2   + (size_t)ROWS * DMODEL;         // 16384*192
    _Float16* Ah = (_Float16*)(ftmp + (size_t)ROWS * PREDLEN); // 16384*1024 halves max
    _Float16* Wh = Ah + (size_t)ROWS * DINNER;                 // 2048*512 halves max

    auto convert = [&](const float* src, _Float16* dst, int count) {
        f32_to_f16_kernel<<<dim3(count / 1024), 256, 0, stream>>>(src, dst);
    };
    auto gemm = [&](const _Float16* A, int lda, const float* Wf, const float* bias,
                    float* C, int ldc, int M, int N, int K, int mode) {
        convert(Wf, Wh, N * K);                 // weights are dense: ldw == K
        dim3 grid(N / 64, M / 64);
        gemm_wmma_kernel<<<grid, 128, 0, stream>>>(A, lda, Wh, K, bias, C, ldc, K, mode);
    };
    auto ew = [&](int count) { return dim3(count / 256); };

    // one mamba block; src already f16 in Ah; slice s = dir*NLAYERS + layer
    auto run_block = [&](int s, float* dst) {
        const float* Win  = w_in   + (size_t)s * 2 * DINNER * DMODEL;
        const float* cw   = conv_w + (size_t)s * DINNER * DCONV;
        const float* cb   = conv_b + (size_t)s * DINNER;
        const float* Wxp  = w_xp   + (size_t)s * 64 * DINNER;
        const float* Wdt  = w_dt   + (size_t)s * DINNER * DTRANK;
        const float* bdt  = b_dt   + (size_t)s * DINNER;
        const float* Alog = a_log  + (size_t)s * DINNER * DSTATE;
        const float* Dp   = d_skip + (size_t)s * DINNER;
        const float* Wout = w_out  + (size_t)s * DMODEL * DINNER;

        gemm(Ah, DMODEL, Win, nullptr, xz, 2 * DINNER, ROWS, 2 * DINNER, DMODEL, 0);
        conv_silu_kernel<<<ew(ROWS * DINNER), 256, 0, stream>>>(xz, cw, cb, xcv);
        convert(xcv, Ah, ROWS * DINNER);
        gemm(Ah, DINNER, Wxp, nullptr, xdbl, 64, ROWS, 64, DINNER, 0);
        convert(xdbl, Ah, ROWS * 64);
        gemm(Ah, 64, Wdt, bdt, dtb, DINNER, ROWS, DINNER, DTRANK, 2 /*bias+softplus*/);
        scan_kernel<<<dim3(DINNER / 256, BATCH), 256, 0, stream>>>(
            dtb, xcv, xdbl, Alog, Dp, ybuf);
        gate_kernel<<<ew(ROWS * DINNER), 256, 0, stream>>>(ybuf, xz, xcv);
        convert(xcv, Ah, ROWS * DINNER);
        gemm(Ah, DINNER, Wout, nullptr, dst, DMODEL, ROWS, DMODEL, DINNER, 0);
    };

    // embedding: emb = inputs^T @ w1^T + b1   (A operand built directly as f16)
    transpose_in_f16_kernel<<<ew(ROWS * NUMLEN), 256, 0, stream>>>(inputs, Ah);
    gemm(Ah, NUMLEN, w1, b1, emb, DMODEL, ROWS, DMODEL, NUMLEN, 1);

    for (int layer = 0; layer < NLAYERS; ++layer) {
        convert(emb, Ah, ROWS * DMODEL);                               // forward dir
        run_block(0 * NLAYERS + layer, r1);
        flip_f16_kernel<<<ew(ROWS * (DMODEL / 4)), 256, 0, stream>>>(emb, Ah);
        run_block(1 * NLAYERS + layer, r2);                            // reverse dir
        combine_kernel<<<ew(ROWS * DMODEL), 256, 0, stream>>>(r1, r2, emb);
    }

    // head: ftmp = emb @ w2^T + b2 ; then (B,L,P) -> (B,P,L)
    convert(emb, Ah, ROWS * DMODEL);
    gemm(Ah, DMODEL, w2, b2, ftmp, PREDLEN, ROWS, PREDLEN, DMODEL, 1);
    transpose_out_kernel<<<ew(BATCH * PREDLEN * SEQL), 256, 0, stream>>>(ftmp, out);
}
